// MulticastAwareGAT_80642305950414
// MI455X (gfx1250) — matrix operand order repaired and checked by reading, not compiled, not run on hardware
//
#include <hip/hip_runtime.h>

// ---------------------------------------------------------------------------
// MulticastAwareGAT forward for MI455X (gfx1250, wave32, WMMA).
// GEMMs: LDS-staged bf16 WMMA (v_wmma_f32_16x16x32_bf16), fully unrolled K.
// Edge softmax/aggregation: L2-resident f32 atomics + ordered-uint atomicMax.
// All attention-vector / gating GEMMs algebraically folded.
// ---------------------------------------------------------------------------

#define NN 50000
#define NE 800000
#define E2 (NE + NN)      // edges + self loops = 850000
#define NF 64
#define EF 32
#define RF 32
#define DD 128
#define HH 4
#define CC 32
#define LL 3
#define ND 64

typedef __attribute__((ext_vector_type(16))) __bf16 v16bf;
typedef __attribute__((ext_vector_type(8)))  float  v8f;

__device__ __forceinline__ unsigned short bfbits(float f)
{
    union { __bf16 h; unsigned short u; } x;
    x.h = (__bf16)f;
    return x.u;
}
__device__ __forceinline__ unsigned packbf2(float a, float b)
{
    return (unsigned)bfbits(a) | ((unsigned)bfbits(b) << 16);
}

// ---------------- LDS-staged WMMA GEMM -------------------------------------
// C[M,N] = act(A[M,K] @ B[K,N] + bias).  K,N compile-time; M runtime.
// Block = 256 threads = 8 waves; block tile = 128 rows x N cols.
// B staged once (bf16, transposed [n][k], padded pitch); A staged per K-step.
template <int K, int N>
__global__ __launch_bounds__(256) void gemm2(const float* __restrict__ A, int lda,
                                             const float* __restrict__ B,
                                             const float* __restrict__ bias,
                                             float* __restrict__ C, int M, int act)
{
    constexpr int PK = K + 8;          // bf16 pitch, keeps 16B alignment, skews banks
    constexpr int NT = N / 16;         // n-tiles per wave
    __shared__ __align__(16) unsigned short sB[N][PK];
    __shared__ __align__(16) unsigned short sA[128][40];   // 32 bf16 + 8 pad

    int t = threadIdx.x;
    int wave = t >> 5, lane = t & 31;
    int mrow = lane & 15, hi = lane >> 4;
    int blockRow = blockIdx.x * 128;
    int tm = wave * 16;

    // ---- stage B: global f32 [K][N] -> LDS bf16 [n][k] ----
    for (int f = t; f < K * N / 4; f += 256) {
        int k  = f / (N / 4);
        int n4 = (f % (N / 4)) * 4;
        const float4 b4 = *(const float4*)(B + (size_t)k * N + n4);
        sB[n4 + 0][k] = bfbits(b4.x);
        sB[n4 + 1][k] = bfbits(b4.y);
        sB[n4 + 2][k] = bfbits(b4.z);
        sB[n4 + 3][k] = bfbits(b4.w);
    }

    v8f acc[NT];
    #pragma unroll
    for (int j = 0; j < NT; ++j) {
        v8f z = {0.f, 0.f, 0.f, 0.f, 0.f, 0.f, 0.f, 0.f};
        acc[j] = z;
    }

    #pragma unroll
    for (int k0 = 0; k0 < K; k0 += 32) {
        __syncthreads();   // guards B staging (first iter) / previous compute
        // ---- stage A chunk: rows [blockRow,blockRow+128) x K [k0,k0+32) ----
        #pragma unroll
        for (int i = 0; i < 4; ++i) {
            int f  = t + i * 256;                 // 1024 float4s
            int r  = f >> 3;
            int c4 = (f & 7) * 4;
            int rg = blockRow + r;
            float4 a4 = make_float4(0.f, 0.f, 0.f, 0.f);
            if (rg < M) a4 = *(const float4*)(A + (size_t)rg * lda + k0 + c4);
            uint2 p;
            p.x = packbf2(a4.x, a4.y);
            p.y = packbf2(a4.z, a4.w);
            *(uint2*)((char*)&sA[r][0] + c4 * 2) = p;
        }
        if (k0 + 32 < K)
            __builtin_prefetch(A + (size_t)(blockRow + (t >> 1)) * lda + k0 + 32, 0, 1);
        __syncthreads();
        // ---- A fragment: two contiguous 16B runs per lane (ISA layout) ----
        union { v16bf v; uint4 q[2]; } af;
        af.q[0] = *(const uint4*)(&sA[tm + mrow][hi * 8]);
        af.q[1] = *(const uint4*)(&sA[tm + mrow][16 + hi * 8]);
        #pragma unroll
        for (int j = 0; j < NT; ++j) {
            union { v16bf v; uint4 q[2]; } bf;
            const unsigned short* bp = &sB[j * 16 + mrow][k0 + hi * 8];
            bf.q[0] = *(const uint4*)(bp);
            bf.q[1] = *(const uint4*)(bp + 16);
            acc[j] = __builtin_amdgcn_wmma_f32_16x16x32_bf16(false, af.v, false, bf.v,
                                                             (short)0, acc[j], false, false);
        }
    }

    // ---- epilogue ----
    #pragma unroll
    for (int j = 0; j < NT; ++j) {
        int n = j * 16 + mrow;
        float bv = (bias != nullptr) ? bias[n] : 0.f;
        #pragma unroll
        for (int r = 0; r < 8; ++r) {
            int m = blockRow + tm + r + hi * 8;
            if (m < M) {
                float v = acc[j][r] + bv;
                if (act) v = fmaxf(v, 0.f);
                C[(size_t)m * N + n] = v;
            }
        }
    }
}

// ---------------- utility fills --------------------------------------------
__global__ void fill_f32(float* p, long n, float v)
{
    long i = blockIdx.x * (long)blockDim.x + threadIdx.x;
    if (i < n) p[i] = v;
}
__global__ void fill_u32(unsigned* p, long n, unsigned v)
{
    long i = blockIdx.x * (long)blockDim.x + threadIdx.x;
    if (i < n) p[i] = v;
}

// ordered-uint encoding for float atomicMax
__device__ __forceinline__ unsigned fenc(float f)
{
    unsigned u = __float_as_uint(f);
    return (u & 0x80000000u) ? ~u : (u | 0x80000000u);
}
__device__ __forceinline__ float fdec(unsigned u)
{
    unsigned b = (u & 0x80000000u) ? (u & 0x7FFFFFFFu) : ~u;
    return __uint_as_float(b);
}

// ---------------- tiny constant-folding kernel -----------------------------
__global__ void prep_kernel(const float* __restrict__ req,
                            const float* __restrict__ mod_rW1, const float* __restrict__ mod_rb1,
                            const float* __restrict__ mod_rW2, const float* __restrict__ mod_rb2,
                            const float* __restrict__ mod_sW,  const float* __restrict__ mod_sb,
                            const float* __restrict__ mod_dW,  const float* __restrict__ mod_db,
                            const float* __restrict__ mod_aW,  const float* __restrict__ mod_ab,
                            const float* __restrict__ gat_eW,  const float* __restrict__ gat_ae,
                            const float* __restrict__ edge_W,  const float* __restrict__ edge_b,
                            float* c_modv, float* c_modc, float* c_ewfold, float* c_ebfold)
{
    __shared__ float rf1[DD], rfv[DD], weff[DD][HH];
    int t = threadIdx.x;
    for (int l = 0; l < LL; ++l) {
        const float* rW1 = mod_rW1 + (size_t)l * RF * DD;
        const float* rW2 = mod_rW2 + (size_t)l * DD * DD;
        if (t < DD) {
            float s = mod_rb1[l * DD + t];
            for (int i = 0; i < RF; ++i) s += req[i] * rW1[i * DD + t];
            rf1[t] = fmaxf(s, 0.f);
        }
        __syncthreads();
        if (t < DD) {
            float s = mod_rb2[l * DD + t];
            for (int i = 0; i < DD; ++i) s += rf1[i] * rW2[i * DD + t];
            rfv[t] = s;
        }
        __syncthreads();
        const float* aW = mod_aW + (size_t)l * 3 * DD;   // [384,1]
        if (t < DD) {
            const float* sW = mod_sW + (size_t)l * DD * DD + (size_t)t * DD;
            const float* dW = mod_dW + (size_t)l * DD * DD + (size_t)t * DD;
            float s = 0.f;
            for (int j = 0; j < DD; ++j) s += sW[j] * aW[j] + dW[j] * aW[DD + j];
            c_modv[l * DD + t] = s;
        }
        for (int idx = t; idx < DD * HH; idx += blockDim.x) {
            int k = idx >> 2, h = idx & 3;
            const float* eW = gat_eW + (size_t)l * DD * DD + (size_t)k * DD + h * CC;
            const float* ae = gat_ae + (size_t)l * HH * CC + h * CC;
            float s = 0.f;
            for (int c = 0; c < CC; ++c) s += eW[c] * ae[c];
            weff[k][h] = s;
        }
        __syncthreads();
        for (int idx = t; idx < EF * HH; idx += blockDim.x) {
            int i = idx >> 2, h = idx & 3;
            const float* eW = edge_W + (size_t)i * DD;
            float s = 0.f;
            for (int k = 0; k < DD; ++k) s += eW[k] * weff[k][h];
            c_ewfold[(l * EF + i) * HH + h] = s;
        }
        if (t < HH) {
            float s = 0.f;
            for (int k = 0; k < DD; ++k) s += edge_b[k] * weff[k][t];
            c_ebfold[l * HH + t] = s;
        }
        if (t == 0) {
            float s = mod_ab[l];
            for (int j = 0; j < DD; ++j)
                s += mod_sb[l * DD + j] * aW[j] + mod_db[l * DD + j] * aW[DD + j]
                   + rfv[j] * aW[2 * DD + j];
            c_modc[l] = s;
        }
        __syncthreads();
    }
}

// ---------------- degree + incoming edge-attr sums -------------------------
__global__ void deg_attr_kernel(const int* __restrict__ ei, const float* __restrict__ eattr,
                                float* deg, float* attr_sum)
{
    int e = blockIdx.x * blockDim.x + threadIdx.x;
    if (e >= NE) return;
    int d = ei[NE + e];
    atomicAdd(deg + d, 1.f);
    const float* a = eattr + (size_t)e * EF;
    float* s = attr_sum + (size_t)d * EF;
    for (int c = 0; c < EF; ++c) atomicAdd(s + c, a[c]);
}

// ---------------- per-node attention projections ---------------------------
__global__ void asrc_adst_kernel(const float* __restrict__ xs,
                                 const float* __restrict__ as, const float* __restrict__ ad,
                                 float* a_src, float* a_dst)
{
    int idx = blockIdx.x * blockDim.x + threadIdx.x;
    if (idx >= NN * HH) return;
    int n = idx >> 2, h = idx & 3;
    const float* row = xs + (size_t)n * DD + h * CC;
    float s1 = 0.f, s2 = 0.f;
    for (int c = 0; c < CC; ++c) {
        float v = row[c];
        s1 += v * as[h * CC + c];
        s2 += v * ad[h * CC + c];
    }
    a_src[idx] = s1;
    a_dst[idx] = s2;
}

// ---------------- edge pass 1: alpha + segment max -------------------------
__global__ void edge_alpha_kernel(const int* __restrict__ ei, const float* __restrict__ eattr,
                                  const float* __restrict__ deg, const float* __restrict__ attr_sum,
                                  const float* __restrict__ a_src, const float* __restrict__ a_dst,
                                  const float* __restrict__ ewfold, const float* __restrict__ ebfold,
                                  float* alpha, unsigned* amax)
{
    __shared__ float s_ew[EF * HH];
    __shared__ float s_eb[HH];
    int t = threadIdx.x;
    if (t < EF * HH) s_ew[t] = ewfold[t];
    if (t < HH) s_eb[t] = ebfold[t];
    __syncthreads();
    long e = blockIdx.x * (long)blockDim.x + t;
    if (e >= E2) return;
    int s, d;
    float ae[HH];
    if (e < NE) {
        s = ei[e]; d = ei[NE + e];
        const float* a = eattr + e * (long)EF;
        for (int h = 0; h < HH; ++h) {
            float acc = s_eb[h];
            for (int i = 0; i < EF; ++i) acc += a[i] * s_ew[i * HH + h];
            ae[h] = acc;
        }
    } else {
        int n = (int)(e - NE); s = n; d = n;   // self loop, mean edge feature
        float inv = 1.f / fmaxf(deg[n], 1.f);
        const float* a = attr_sum + (size_t)n * EF;
        for (int h = 0; h < HH; ++h) {
            float acc = s_eb[h];
            for (int i = 0; i < EF; ++i) acc += a[i] * inv * s_ew[i * HH + h];
            ae[h] = acc;
        }
    }
    for (int h = 0; h < HH; ++h) {
        float al = a_src[s * HH + h] + a_dst[d * HH + h] + ae[h];
        al = (al > 0.f) ? al : 0.2f * al;      // leaky_relu 0.2
        alpha[e * HH + h] = al;
        atomicMax(amax + d * HH + h, fenc(al));
    }
}

// ---------------- edge pass 2: exp + segment sum ---------------------------
__global__ void edge_exp_kernel(const int* __restrict__ ei, const unsigned* __restrict__ amax,
                                float* alpha, float* denom)
{
    long e = blockIdx.x * (long)blockDim.x + threadIdx.x;
    if (e >= E2) return;
    int d = (e < NE) ? ei[NE + e] : (int)(e - NE);
    for (int h = 0; h < HH; ++h) {
        float ex = __expf(alpha[e * HH + h] - fdec(amax[d * HH + h]));
        alpha[e * HH + h] = ex;
        atomicAdd(denom + d * HH + h, ex);
    }
}

// ---------------- edge pass 3: weighted aggregation (wave per edge) --------
__global__ void edge_aggr_kernel(const int* __restrict__ ei, const float* __restrict__ alpha,
                                 const float* __restrict__ denom, const float* __restrict__ xs,
                                 float* out)
{
    int lane = threadIdx.x & 31;
    long e = (blockIdx.x * (long)blockDim.x + threadIdx.x) >> 5;
    if (e >= E2) return;
    int s, d;
    if (e < NE) { s = ei[e]; d = ei[NE + e]; }
    else        { s = (int)(e - NE); d = s; }
    int h = lane >> 3;                               // 4 ch/lane, 8 lanes/head
    float att = alpha[e * HH + h] / (denom[d * HH + h] + 1e-16f);
    const float4 v = *(const float4*)(xs + (size_t)s * DD + lane * 4);
    float* o = out + (size_t)d * DD + lane * 4;
    atomicAdd(o + 0, v.x * att);
    atomicAdd(o + 1, v.y * att);
    atomicAdd(o + 2, v.z * att);
    atomicAdd(o + 3, v.w * att);
}

// ---------------- gate + residual + layernorm (wave per node) --------------
__global__ void mod_ln_kernel(const float* __restrict__ x_in, const float* __restrict__ agg,
                              const float* __restrict__ gatb, const float* __restrict__ modv,
                              const float* __restrict__ modc_p,
                              const float* __restrict__ ln_g, const float* __restrict__ ln_b,
                              float* x_out)
{
    int lane = threadIdx.x & 31;
    long n = (blockIdx.x * (long)blockDim.x + threadIdx.x) >> 5;
    if (n >= NN) return;
    const float4 xr = *(const float4*)(x_in + n * (long)DD + lane * 4);
    const float4 mv = *(const float4*)(modv + lane * 4);
    float dotp = xr.x * mv.x + xr.y * mv.y + xr.z * mv.z + xr.w * mv.w;
    for (int o = 16; o >= 1; o >>= 1) dotp += __shfl_xor(dotp, o, 32);
    float w = 1.f / (1.f + __expf(-(dotp + modc_p[0])));
    const float4 ag = *(const float4*)(agg + n * (long)DD + lane * 4);
    const float4 gb = *(const float4*)(gatb + lane * 4);
    float t0 = xr.x + (ag.x + gb.x) * w;
    float t1 = xr.y + (ag.y + gb.y) * w;
    float t2 = xr.z + (ag.z + gb.z) * w;
    float t3 = xr.w + (ag.w + gb.w) * w;
    float m = t0 + t1 + t2 + t3;
    for (int o = 16; o >= 1; o >>= 1) m += __shfl_xor(m, o, 32);
    m *= (1.f / DD);
    float v0 = t0 - m, v1 = t1 - m, v2 = t2 - m, v3 = t3 - m;
    float var = v0 * v0 + v1 * v1 + v2 * v2 + v3 * v3;
    for (int o = 16; o >= 1; o >>= 1) var += __shfl_xor(var, o, 32);
    var *= (1.f / DD);
    float inv = rsqrtf(var + 1e-5f);
    const float4 g = *(const float4*)(ln_g + lane * 4);
    const float4 b = *(const float4*)(ln_b + lane * 4);
    float4 r;
    r.x = v0 * inv * g.x + b.x;
    r.y = v1 * inv * g.y + b.y;
    r.z = v2 * inv * g.z + b.z;
    r.w = v3 * inv * g.w + b.w;
    *(float4*)(x_out + n * (long)DD + lane * 4) = r;
}

// ---------------- SetTransformer (tiny, M=64) ------------------------------
__global__ void st_enc_kernel(const float* __restrict__ x0, const int* __restrict__ dest,
                              const float* __restrict__ eW1, const float* __restrict__ eb1,
                              const float* __restrict__ eW2, const float* __restrict__ eb2,
                              float* enc)
{
    __shared__ float h1[ND][DD];
    int t = threadIdx.x;
    for (int idx = t; idx < ND * DD; idx += blockDim.x) {
        int m = idx >> 7, j = idx & 127;
        const float* xr = x0 + (size_t)dest[m] * DD;
        float s = eb1[j];
        for (int i = 0; i < DD; ++i) s += xr[i] * eW1[i * DD + j];
        h1[m][j] = fmaxf(s, 0.f);
    }
    __syncthreads();
    for (int idx = t; idx < ND * DD; idx += blockDim.x) {
        int m = idx >> 7, j = idx & 127;
        float s = eb2[j];
        for (int i = 0; i < DD; ++i) s += h1[m][i] * eW2[i * DD + j];
        enc[idx] = s;
    }
}

__global__ void st_qkv_kernel(const float* __restrict__ enc, const float* __restrict__ qkvW,
                              const float* __restrict__ qkvb, float* qkv)
{
    int t = threadIdx.x;
    for (int idx = t; idx < ND * 3 * DD; idx += blockDim.x) {
        int m = idx / (3 * DD), j = idx % (3 * DD);
        const float* er = enc + (size_t)m * DD;
        float s = qkvb[j];
        for (int i = 0; i < DD; ++i) s += er[i] * qkvW[i * 3 * DD + j];
        qkv[idx] = s;
    }
}

__global__ void st_attn_kernel(const float* __restrict__ qkv, float* attout)
{
    __shared__ float sc[ND][ND];
    int t = threadIdx.x;
    for (int h = 0; h < HH; ++h) {
        for (int idx = t; idx < ND * ND; idx += blockDim.x) {
            int m = idx >> 6, n = idx & 63;
            const float* q = qkv + (size_t)m * 3 * DD + h * CC;
            const float* k = qkv + (size_t)n * 3 * DD + DD + h * CC;
            float s = 0.f;
            for (int c = 0; c < CC; ++c) s += q[c] * k[c];
            sc[m][n] = s * 0.17677669529663687f;   // 1/sqrt(32)
        }
        __syncthreads();
        if (t < ND) {
            float mx = -1e30f;
            for (int n = 0; n < ND; ++n) mx = fmaxf(mx, sc[t][n]);
            float sum = 0.f;
            for (int n = 0; n < ND; ++n) { float e = __expf(sc[t][n] - mx); sc[t][n] = e; sum += e; }
            float inv = 1.f / sum;
            for (int n = 0; n < ND; ++n) sc[t][n] *= inv;
        }
        __syncthreads();
        for (int idx = t; idx < ND * CC; idx += blockDim.x) {
            int m = idx >> 5, c = idx & 31;
            float s = 0.f;
            for (int n = 0; n < ND; ++n) s += sc[m][n] * qkv[(size_t)n * 3 * DD + 2 * DD + h * CC + c];
            attout[(size_t)m * DD + h * CC + c] = s;
        }
        __syncthreads();
    }
}

__global__ void st_final_kernel(const float* __restrict__ attout,
                                const float* __restrict__ oW, const float* __restrict__ ob,
                                const float* __restrict__ aW1, const float* __restrict__ ab1,
                                const float* __restrict__ aW2, const float* __restrict__ ab2,
                                const float* __restrict__ pr_W1, const float* __restrict__ pr_b1,
                                float* dest_out, float* dest_ws, float* prb1_eff)
{
    __shared__ float colsum[DD], sv[DD], t1[DD], ds[DD];
    int t = threadIdx.x;
    if (t < DD) {
        float c = 0.f;
        for (int m = 0; m < ND; ++m) c += attout[(size_t)m * DD + t];
        colsum[t] = c;
    }
    __syncthreads();
    if (t < DD) {
        float v = (float)ND * ob[t];                 // sum_m(o@Wo + ob)
        for (int k = 0; k < DD; ++k) v += colsum[k] * oW[k * DD + t];
        sv[t] = v;
    }
    __syncthreads();
    if (t < DD) {
        float v = ab1[t];
        for (int k = 0; k < DD; ++k) v += sv[k] * aW1[k * DD + t];
        t1[t] = fmaxf(v, 0.f);
    }
    __syncthreads();
    if (t < DD) {
        float v = ab2[t];
        for (int k = 0; k < DD; ++k) v += t1[k] * aW2[k * DD + t];
        ds[t] = v;
        dest_out[t] = v;
        dest_ws[t] = v;
    }
    __syncthreads();
    if (t < DD) {                                    // fold dest_set into pr_b1
        float v = pr_b1[t];
        for (int k = 0; k < DD; ++k) v += ds[k] * pr_W1[(size_t)(DD + k) * DD + t];
        prb1_eff[t] = v;
    }
}

// ---------------- final score ---------------------------------------------
__global__ void score_kernel(const float* __restrict__ h2, const float* __restrict__ W3,
                             const float* __restrict__ b3, float* scores)
{
    int n = blockIdx.x * blockDim.x + threadIdx.x;
    if (n >= NN) return;
    const float* r = h2 + (size_t)n * 64;
    float s = b3[0];
    for (int i = 0; i < 64; ++i) s += r[i] * W3[i];
    scores[n] = 1.f / (1.f + __expf(-s));
}

// ---------------------------------------------------------------------------
static inline int cdiv(long a, long b) { return (int)((a + b - 1) / b); }

extern "C" void kernel_launch(void* const* d_in, const int* in_sizes, int n_in,
                              void* d_out, int out_size, void* d_ws, size_t ws_size,
                              hipStream_t stream)
{
    (void)in_sizes; (void)n_in; (void)out_size; (void)ws_size;
    int i = 0;
    const float* x_feat   = (const float*)d_in[i++];   // 0
    const int*   eidx     = (const int*)  d_in[i++];   // 1
    const float* eattr    = (const float*)d_in[i++];   // 2
    const float* req      = (const float*)d_in[i++];   // 3
    const int*   dest     = (const int*)  d_in[i++];   // 4
    const float* node_W   = (const float*)d_in[i++];
    const float* node_b   = (const float*)d_in[i++];
    const float* edge_W   = (const float*)d_in[i++];
    const float* edge_b   = (const float*)d_in[i++];
    const float* st_eW1   = (const float*)d_in[i++];
    const float* st_eb1   = (const float*)d_in[i++];
    const float* st_eW2   = (const float*)d_in[i++];
    const float* st_eb2   = (const float*)d_in[i++];
    const float* st_qkvW  = (const float*)d_in[i++];
    const float* st_qkvb  = (const float*)d_in[i++];
    const float* st_oW    = (const float*)d_in[i++];
    const float* st_ob    = (const float*)d_in[i++];
    const float* st_aW1   = (const float*)d_in[i++];
    const float* st_ab1   = (const float*)d_in[i++];
    const float* st_aW2   = (const float*)d_in[i++];
    const float* st_ab2   = (const float*)d_in[i++];
    const float* gat_W    = (const float*)d_in[i++];
    const float* gat_as   = (const float*)d_in[i++];
    const float* gat_ad   = (const float*)d_in[i++];
    const float* gat_eW   = (const float*)d_in[i++];
    const float* gat_ae   = (const float*)d_in[i++];
    const float* gat_b    = (const float*)d_in[i++];
    const float* mod_sW   = (const float*)d_in[i++];
    const float* mod_sb   = (const float*)d_in[i++];
    const float* mod_dW   = (const float*)d_in[i++];
    const float* mod_db   = (const float*)d_in[i++];
    const float* mod_rW1  = (const float*)d_in[i++];
    const float* mod_rb1  = (const float*)d_in[i++];
    const float* mod_rW2  = (const float*)d_in[i++];
    const float* mod_rb2  = (const float*)d_in[i++];
    const float* mod_aW   = (const float*)d_in[i++];
    const float* mod_ab   = (const float*)d_in[i++];
    const float* ln_g     = (const float*)d_in[i++];
    const float* ln_b     = (const float*)d_in[i++];
    const float* pr_W1    = (const float*)d_in[i++];
    const float* pr_b1    = (const float*)d_in[i++];
    const float* pr_W2    = (const float*)d_in[i++];
    const float* pr_b2    = (const float*)d_in[i++];
    const float* pr_W3    = (const float*)d_in[i++];
    const float* pr_b3    = (const float*)d_in[i++];

    // output regions: x [NN*DD], dest_set [DD], scores [NN]
    float* out_x      = (float*)d_out;
    float* out_dest   = out_x + (size_t)NN * DD;
    float* out_scores = out_dest + DD;

    // workspace carve-out
    char* w = (char*)d_ws;
    auto carve = [&](size_t bytes) -> float* {
        float* p = (float*)w;
        w += (bytes + 255) & ~(size_t)255;
        return p;
    };
    float*    x_cur    = carve((size_t)NN * DD * 4);
    float*    xs       = carve((size_t)NN * DD * 4);
    float*    agg      = carve((size_t)NN * DD * 4);
    float*    alpha    = carve((size_t)E2 * HH * 4);     // reused as h2 later
    float*    attr_sum = carve((size_t)NN * EF * 4);
    float*    deg      = carve((size_t)NN * 4);
    float*    a_src    = carve((size_t)NN * HH * 4);
    float*    a_dst    = carve((size_t)NN * HH * 4);
    unsigned* amax     = (unsigned*)carve((size_t)NN * HH * 4);
    float*    denom    = carve((size_t)NN * HH * 4);
    float*    enc      = carve((size_t)ND * DD * 4);
    float*    qkv      = carve((size_t)ND * 3 * DD * 4);
    float*    attout   = carve((size_t)ND * DD * 4);
    float*    dest_ws  = carve(DD * 4);
    float*    prb1_eff = carve(DD * 4);
    float*    c_modv   = carve(LL * DD * 4);
    float*    c_modc   = carve(LL * 4);
    float*    c_ewfold = carve(LL * EF * HH * 4);
    float*    c_ebfold = carve(LL * HH * 4);
    float*    h1 = xs;        // free after layers
    float*    h2 = alpha;     // 50000*64 floats <= E2*HH floats

    const int TB = 256;
    const int gemmBlocks = cdiv(NN, 128);   // 128-row block tiles

    // ---- node embedding: x_cur = x_feat @ node_W + node_b (WMMA) ----
    gemm2<NF, DD><<<gemmBlocks, TB, 0, stream>>>(x_feat, NF, node_W, node_b, x_cur, NN, 0);

    // ---- SetTransformer over destination nodes ----
    st_enc_kernel<<<1, 256, 0, stream>>>(x_cur, dest, st_eW1, st_eb1, st_eW2, st_eb2, enc);
    st_qkv_kernel<<<1, 256, 0, stream>>>(enc, st_qkvW, st_qkvb, qkv);
    st_attn_kernel<<<1, 256, 0, stream>>>(qkv, attout);
    st_final_kernel<<<1, 128, 0, stream>>>(attout, st_oW, st_ob, st_aW1, st_ab1,
                                           st_aW2, st_ab2, pr_W1, pr_b1,
                                           out_dest, dest_ws, prb1_eff);

    // ---- constant folding ----
    prep_kernel<<<1, 256, 0, stream>>>(req, mod_rW1, mod_rb1, mod_rW2, mod_rb2,
                                       mod_sW, mod_sb, mod_dW, mod_db, mod_aW, mod_ab,
                                       gat_eW, gat_ae, edge_W, edge_b,
                                       c_modv, c_modc, c_ewfold, c_ebfold);

    // ---- degree + incoming edge-attr sums (self-loop mean features) ----
    fill_f32<<<cdiv(NN, TB), TB, 0, stream>>>(deg, NN, 0.f);
    fill_f32<<<cdiv((long)NN * EF, TB), TB, 0, stream>>>(attr_sum, (long)NN * EF, 0.f);
    deg_attr_kernel<<<cdiv(NE, TB), TB, 0, stream>>>(eidx, eattr, deg, attr_sum);

    // ---- GAT layers ----
    const float* x_in = x_cur;
    for (int l = 0; l < LL; ++l) {
        gemm2<DD, DD><<<gemmBlocks, TB, 0, stream>>>(
            x_in, DD, gat_W + (size_t)l * DD * DD, nullptr, xs, NN, 0);
        asrc_adst_kernel<<<cdiv((long)NN * HH, TB), TB, 0, stream>>>(
            xs, gat_as + (size_t)l * HH * CC, gat_ad + (size_t)l * HH * CC, a_src, a_dst);
        fill_u32<<<cdiv((long)NN * HH, TB), TB, 0, stream>>>(amax, (long)NN * HH, 0u);
        fill_f32<<<cdiv((long)NN * HH, TB), TB, 0, stream>>>(denom, (long)NN * HH, 0.f);
        fill_f32<<<cdiv((long)NN * DD, TB), TB, 0, stream>>>(agg, (long)NN * DD, 0.f);
        edge_alpha_kernel<<<cdiv(E2, TB), TB, 0, stream>>>(
            eidx, eattr, deg, attr_sum, a_src, a_dst,
            c_ewfold + (size_t)l * EF * HH, c_ebfold + (size_t)l * HH, alpha, amax);
        edge_exp_kernel<<<cdiv(E2, TB), TB, 0, stream>>>(eidx, amax, alpha, denom);
        edge_aggr_kernel<<<cdiv((long)E2 * 32, TB), TB, 0, stream>>>(eidx, alpha, denom, xs, agg);
        float* x_out = (l == LL - 1) ? out_x : x_cur;
        mod_ln_kernel<<<cdiv((long)NN * 32, TB), TB, 0, stream>>>(
            x_in, agg, gat_b + (size_t)l * DD, c_modv + (size_t)l * DD, c_modc + l,
            ln_g + (size_t)l * DD, ln_b + (size_t)l * DD, x_out);
        x_in = x_out;
    }

    // ---- predictor (dest_set folded into prb1_eff) ----
    gemm2<DD, DD><<<gemmBlocks, TB, 0, stream>>>(out_x, DD, pr_W1, prb1_eff, h1, NN, 1);
    gemm2<DD, 64><<<gemmBlocks, TB, 0, stream>>>(h1, DD, pr_W2, pr_b2, h2, NN, 1);
    score_kernel<<<cdiv(NN, TB), TB, 0, stream>>>(h2, pr_W3, pr_b3, out_scores);
}
